// EvoformerBlockSS_28733331210463
// MI455X (gfx1250) — matrix operand order repaired and checked by reading, not compiled, not run on hardware
//
#include <hip/hip_runtime.h>

// ---------------------------------------------------------------------------
// EvoformerBlock forward for MI455X (gfx1250, wave32, WMMA).
// All matmuls route through one strided-batched bf16 WMMA GEMM kernel
// (v_wmma_f32_16x16x32_bf16, fp32 accum): 64x64 block tile, 4 waves,
// each wave 2x2 16x16 accumulators -> 4 WMMAs per K-step with fragment reuse.
// LDS tiles are stored as packed bf16 *pairs* in the exact dword layout the
// WMMA fragments need, so every fragment dword is one aligned ds_load_b32.
// VEC=1: float4 (b128) staging for row-major operands; VEC=0: generic strides.
// OOB handling is branch-free: clamp 32-bit indices, mask loaded values.
// Next K-tile is prefetched (global_prefetch_b8) while the current one runs.
// ---------------------------------------------------------------------------

typedef __attribute__((ext_vector_type(16))) __bf16          v16bf;
typedef __attribute__((ext_vector_type(8)))  unsigned        v8u;
typedef __attribute__((ext_vector_type(8)))  float           v8f;

constexpr int L  = 320;
constexpr int CS = 1024;
constexpr int CZ = 128;
constexpr int HS = 32;   // seq heads, hd = CS/HS = 32
constexpr int HP = 4;    // pair heads, hd = CZ/HP = 32
constexpr int HD = 32;
constexpr int CI = 32;       // i-chunk for triangle attention
constexpr int RCHUNK = 10240; // row chunk for pair MLP

__device__ __forceinline__ unsigned short f2bf(float f) {
  union { float f; unsigned u; } v; v.f = f;
  unsigned u = v.u + 0x7FFFu + ((v.u >> 16) & 1u);   // round-to-nearest-even
  return (unsigned short)(u >> 16);
}
__device__ __forceinline__ unsigned pk2bf(float x, float y) {
  return (unsigned)f2bf(x) | ((unsigned)f2bf(y) << 16);
}

// ---------------------------------------------------------------------------
// Generic strided batched GEMM:  C[m,n] = sum_k A[m,k]*B[k,n] (+bias[n]) (+act)
// batch z decomposes as (outer, inner) with independent strides; inner count
// is a power of two -> shift/mask decode.
// VEC=1: ldaK==1, ldbN==1, ldcN==1 assumed; K multiple of 32; N multiple of 4;
//        all row strides multiples of 4 floats (16B) -> float4 staging legal.
// VEC=0: fully generic element strides, scalar branch-free staging.
// ---------------------------------------------------------------------------
struct GemmArgs {
  const float* A; const float* B; const float* bias; float* C;
  int M, N, K, batchShift;   // inner batch count == 1 << batchShift
  long long ldaM, ldaK, ldbK, ldbN, ldcM, ldcN;
  long long sAo, sAi, sBo, sBi, sCo, sCi;
  int fuse;   // 0 none, 1 relu, 2 sigmoid
};

template <int VEC>
__global__ __launch_bounds__(128) void wmma_gemm_kernel(GemmArgs g) {
  const int bo = (int)blockIdx.z >> g.batchShift;
  const int bi = (int)blockIdx.z & ((1 << g.batchShift) - 1);
  const float* A = g.A + (long long)bo * g.sAo + (long long)bi * g.sAi;
  const float* B = g.B + (long long)bo * g.sBo + (long long)bi * g.sBi;
  float*       C = g.C + (long long)bo * g.sCo + (long long)bi * g.sCi;

  const int m0 = blockIdx.x * 64;
  const int n0 = blockIdx.y * 64;

  // As32[m][kd]  : dword kd packs bf16 pair (K=2kd, K=2kd+1) of row m
  // Bs32[kp][n]  : dword packs bf16 pair (K=2kp, K=2kp+1) of column n
  __shared__ unsigned As32[64][17];   // 16 dwords + 1 pad
  __shared__ unsigned Bs32[16][65];   // 64 dwords + 1 pad

  const int tid  = threadIdx.x;
  const int wave = tid >> 5;
  const int lane = tid & 31;
  const int wm = (wave & 1) * 32;   // wave's 32x32 sub-tile origin
  const int wn = (wave >> 1) * 32;
  const int mr = lane & 15;
  const int kh = lane >> 4;

  v8f c00 = {}, c01 = {}, c10 = {}, c11 = {};

  for (int k0 = 0; k0 < g.K; k0 += 32) {
    // ---- prefetch next K-tile (one global_prefetch_b8 per thread) ----
    {
      const int kn0 = (k0 + 32 < g.K) ? (k0 + 32) : k0;
      if (tid < 64) {                               // A: 64 rows
        const int pr = (m0 + tid < g.M) ? (m0 + tid) : 0;
        __builtin_prefetch(A + (long long)pr * g.ldaM + (long long)kn0 * g.ldaK, 0, 1);
      } else {                                      // B: 32 rows x 2 halves
        const int t2 = tid - 64;
        const int br = t2 >> 1;
        const int bc = (t2 & 1) * 32;
        const int col = (n0 + bc < g.N) ? (n0 + bc) : 0;
        __builtin_prefetch(B + (long long)(kn0 + br) * g.ldbK + (long long)col * g.ldbN, 0, 1);
      }
    }
    if (VEC) {
      // ---- A: 64 rows x 32 cols = 512 float4, 4 per thread ----
#pragma unroll
      for (int i = 0; i < 4; ++i) {
        const int idx = tid + i * 128;        // 0..511
        const int r   = idx >> 3;
        const int c4  = (idx & 7) << 2;
        const bool inb = (m0 + r) < g.M;
        const int row  = inb ? (m0 + r) : 0;  // clamp index, not offset
        float4 v = *(const float4*)(A + (long long)row * g.ldaM + (k0 + c4));
        if (!inb) { v.x = 0.f; v.y = 0.f; v.z = 0.f; v.w = 0.f; }
        As32[r][(c4 >> 1)]     = pk2bf(v.x, v.y);
        As32[r][(c4 >> 1) + 1] = pk2bf(v.z, v.w);
      }
      // ---- B: 16 kpairs x 16 col-groups = 256 tasks, 2 float4 each ----
#pragma unroll
      for (int i = 0; i < 2; ++i) {
        const int idx = tid + i * 128;        // 0..255
        const int kp  = idx >> 4;             // 0..15
        const int c4  = (idx & 15) << 2;      // 0..60
        const bool inb = (n0 + c4) < g.N;     // N multiple of 4
        const int col  = inb ? (n0 + c4) : 0;
        const long long r0 = (long long)(k0 + 2 * kp) * g.ldbK + col;
        float4 u = *(const float4*)(B + r0);
        float4 v = *(const float4*)(B + r0 + g.ldbK);
        if (!inb) { u.x=0.f;u.y=0.f;u.z=0.f;u.w=0.f; v.x=0.f;v.y=0.f;v.z=0.f;v.w=0.f; }
        Bs32[kp][c4]     = pk2bf(u.x, v.x);
        Bs32[kp][c4 + 1] = pk2bf(u.y, v.y);
        Bs32[kp][c4 + 2] = pk2bf(u.z, v.z);
        Bs32[kp][c4 + 3] = pk2bf(u.w, v.w);
      }
    } else {
      // ---- generic strided staging, branch-free OOB ----
      unsigned short* ash = (unsigned short*)&As32[0][0];  // row stride 34 halves
      unsigned short* bsh = (unsigned short*)&Bs32[0][0];  // kpair stride 130 halves
#pragma unroll 4
      for (int t = tid; t < 2048; t += 128) {      // A: 64x32
        const int r = t >> 5, c = t & 31;
        const bool inb = (m0 + r < g.M) && (k0 + c < g.K);
        const int mi = inb ? (m0 + r) : 0;
        const int ki = inb ? (k0 + c) : 0;
        const float v = A[(long long)mi * g.ldaM + (long long)ki * g.ldaK];
        ash[r * 34 + c] = f2bf(inb ? v : 0.f);
      }
#pragma unroll 4
      for (int t = tid; t < 2048; t += 128) {      // B: 32x64
        const int r = t >> 6, c = t & 63;
        const bool inb = (k0 + r < g.K) && (n0 + c < g.N);
        const int ki = inb ? (k0 + r) : 0;
        const int ni = inb ? (n0 + c) : 0;
        const float v = B[(long long)ki * g.ldbK + (long long)ni * g.ldbN];
        bsh[(r >> 1) * 130 + c * 2 + (r & 1)] = f2bf(inb ? v : 0.f);
      }
    }
    __syncthreads();

    // Fragments: every dword is one aligned 32-bit LDS load.
    // A 16x32: lane = M row (0-15), K halves split across lane>>4.
    // B 32x16: lane = N col (0-15), K halves split across lane>>4.
    v8u a0u, a1u, b0u, b1u;
#pragma unroll
    for (int r = 0; r < 8; ++r) {
      const int kd = r + 8 * kh;                           // A dword index
      a0u[r] = As32[wm + mr][kd];
      a1u[r] = As32[wm + 16 + mr][kd];
      const int kp = r + 8 * kh;                           // B kpair index
      b0u[r] = Bs32[kp][wn + mr];
      b1u[r] = Bs32[kp][wn + 16 + mr];
    }
    const v16bf a0 = __builtin_bit_cast(v16bf, a0u);
    const v16bf a1 = __builtin_bit_cast(v16bf, a1u);
    const v16bf b0 = __builtin_bit_cast(v16bf, b0u);
    const v16bf b1 = __builtin_bit_cast(v16bf, b1u);
    c00 = __builtin_amdgcn_wmma_f32_16x16x32_bf16(false, a0, false, b0, (short)0, c00, false, false);
    c01 = __builtin_amdgcn_wmma_f32_16x16x32_bf16(false, a0, false, b1, (short)0, c01, false, false);
    c10 = __builtin_amdgcn_wmma_f32_16x16x32_bf16(false, a1, false, b0, (short)0, c10, false, false);
    c11 = __builtin_amdgcn_wmma_f32_16x16x32_bf16(false, a1, false, b1, (short)0, c11, false, false);
    __syncthreads();
  }

  // epilogue: C/D layout -> lane n = lane&15, rows r + 8*(lane>>4)
  const int cn = lane & 15;
  const int ch = lane >> 4;
  auto store_tile = [&](const v8f& acc, int ti, int tj) {
#pragma unroll
    for (int r = 0; r < 8; ++r) {
      const int gm = m0 + wm + 16 * ti + r + 8 * ch;
      const int gn = n0 + wn + 16 * tj + cn;
      if (gm < g.M && gn < g.N) {
        float v = acc[r];
        if (g.bias) v += g.bias[gn];
        if (g.fuse == 1)      v = fmaxf(v, 0.f);
        else if (g.fuse == 2) v = 1.f / (1.f + __expf(-v));
        const long long off = VEC ? ((long long)gm * g.ldcM + gn)
                                  : ((long long)gm * g.ldcM + (long long)gn * g.ldcN);
        C[off] = v;
      }
    }
  };
  store_tile(c00, 0, 0);
  store_tile(c01, 0, 1);
  store_tile(c10, 1, 0);
  store_tile(c11, 1, 1);
}

// ---------------------------------------------------------------------------
// LayerNorm over last (contiguous) dim D; one block per row.
// ---------------------------------------------------------------------------
__global__ __launch_bounds__(256) void ln_kernel(const float* __restrict__ x,
                                                 const float* __restrict__ gam,
                                                 const float* __restrict__ bet,
                                                 float* __restrict__ y, int D) {
  const long long row = blockIdx.x;
  const float* xr = x + row * (long long)D;
  float*       yr = y + row * (long long)D;
  __shared__ float s1[256], s2[256];
  float sum = 0.f, sq = 0.f;
  for (int i = threadIdx.x; i < D; i += blockDim.x) {
    float v = xr[i]; sum += v; sq += v * v;
  }
  s1[threadIdx.x] = sum; s2[threadIdx.x] = sq;
  __syncthreads();
  for (int o = 128; o > 0; o >>= 1) {
    if ((int)threadIdx.x < o) { s1[threadIdx.x] += s1[threadIdx.x + o];
                                s2[threadIdx.x] += s2[threadIdx.x + o]; }
    __syncthreads();
  }
  const float mean = s1[0] / (float)D;
  const float var  = s2[0] / (float)D - mean * mean;
  const float inv  = rsqrtf(var + 1e-5f);
  for (int i = threadIdx.x; i < D; i += blockDim.x)
    yr[i] = (xr[i] - mean) * inv * gam[i] + bet[i];
}

// ---------------------------------------------------------------------------
// Softmax kernels (rows of length L).
// ---------------------------------------------------------------------------
__global__ __launch_bounds__(256) void seq_softmax_kernel(float* __restrict__ sc,
                                                          const float* __restrict__ bias,
                                                          const float* __restrict__ mask,
                                                          float scale) {
  const int q = blockIdx.x, h = blockIdx.y, tid = threadIdx.x;
  float* row = sc + ((long long)h * L + q) * (long long)L;
  __shared__ float red[256];
  float mx = -1e30f;
  for (int k = tid; k < L; k += 256) {
    float v = row[k] * scale + bias[((long long)q * L + k) * HS + h];
    v = (mask[k] > 0.f) ? v : -1e30f;
    row[k] = v;
    mx = fmaxf(mx, v);
  }
  red[tid] = mx; __syncthreads();
  for (int o = 128; o > 0; o >>= 1) {
    if (tid < o) red[tid] = fmaxf(red[tid], red[tid + o]);
    __syncthreads();
  }
  mx = red[0]; __syncthreads();
  float sum = 0.f;
  for (int k = tid; k < L; k += 256) {
    float e = __expf(row[k] - mx); row[k] = e; sum += e;
  }
  red[tid] = sum; __syncthreads();
  for (int o = 128; o > 0; o >>= 1) {
    if (tid < o) red[tid] += red[tid + o];
    __syncthreads();
  }
  const float inv = 1.f / (red[0] + 1e-20f);
  for (int k = tid; k < L; k += 256) row[k] *= inv;
}

__global__ __launch_bounds__(256) void tri_softmax_kernel(float* __restrict__ sc,
                                                          const float* __restrict__ pm,
                                                          const float* __restrict__ tb,
                                                          int i0, float scale) {
  const int q = blockIdx.x, h = blockIdx.y, li = blockIdx.z, tid = threadIdx.x;
  const int i = i0 + li;
  float* row = sc + (((long long)li * HP + h) * L + q) * (long long)L;
  __shared__ float red[256];
  float mx = -1e30f;
  for (int k = tid; k < L; k += 256) {
    float v = row[k] * scale
            + (pm[(long long)i * L + k] - 1.f) * 1e9f
            + tb[((long long)q * L + k) * HP + h];   // triangle bias at (q,k)
    row[k] = v;
    mx = fmaxf(mx, v);
  }
  red[tid] = mx; __syncthreads();
  for (int o = 128; o > 0; o >>= 1) {
    if (tid < o) red[tid] = fmaxf(red[tid], red[tid + o]);
    __syncthreads();
  }
  mx = red[0]; __syncthreads();
  float sum = 0.f;
  for (int k = tid; k < L; k += 256) {
    float e = __expf(row[k] - mx); row[k] = e; sum += e;
  }
  red[tid] = sum; __syncthreads();
  for (int o = 128; o > 0; o >>= 1) {
    if (tid < o) red[tid] += red[tid + o];
    __syncthreads();
  }
  const float inv = 1.f / (red[0] + 1e-20f);
  for (int k = tid; k < L; k += 256) row[k] *= inv;
}

// ---------------------------------------------------------------------------
// Elementwise helpers (grid-stride).
// ---------------------------------------------------------------------------
__global__ void pm_kernel(const float* __restrict__ mask, float* __restrict__ pm) {
  long long idx = (long long)blockIdx.x * blockDim.x + threadIdx.x;
  long long n = (long long)L * L;
  for (; idx < n; idx += (long long)gridDim.x * blockDim.x)
    pm[idx] = mask[idx / L] * mask[idx % L];
}
__global__ void add_kernel(float* __restrict__ y, const float* __restrict__ a,
                           const float* __restrict__ b, long long n) {
  long long i = (long long)blockIdx.x * blockDim.x + threadIdx.x;
  for (; i < n; i += (long long)gridDim.x * blockDim.x) y[i] = a[i] + b[i];
}
__global__ void add_inplace_kernel(float* __restrict__ y, const float* __restrict__ x,
                                   long long n) {
  long long i = (long long)blockIdx.x * blockDim.x + threadIdx.x;
  for (; i < n; i += (long long)gridDim.x * blockDim.x) y[i] += x[i];
}
__global__ void mul_kernel(float* __restrict__ y, const float* __restrict__ x, long long n) {
  long long i = (long long)blockIdx.x * blockDim.x + threadIdx.x;
  for (; i < n; i += (long long)gridDim.x * blockDim.x) y[i] *= x[i];
}
__global__ void addmul_kernel(float* __restrict__ y, const float* __restrict__ a,
                              const float* __restrict__ b, long long n) {
  long long i = (long long)blockIdx.x * blockDim.x + threadIdx.x;
  for (; i < n; i += (long long)gridDim.x * blockDim.x) y[i] += a[i] * b[i];
}
// out = pm[ij] * g * p   (g already sigmoided by GEMM epilogue)
__global__ void tri_gate_kernel(float* __restrict__ out, const float* __restrict__ g,
                                const float* __restrict__ p, const float* __restrict__ pm,
                                long long n) {
  long long i = (long long)blockIdx.x * blockDim.x + threadIdx.x;
  for (; i < n; i += (long long)gridDim.x * blockDim.x)
    out[i] = pm[i / CZ] * g[i] * p[i];
}
// cat[i,j,c] : c<64 -> q[j,c]*k[i,c] ; c>=64 -> q[j,c-64]-k[i,c-64]
__global__ __launch_bounds__(128) void s2p_build_kernel(const float* __restrict__ proj,
                                                        float* __restrict__ cat) {
  const int i = blockIdx.x, j = blockIdx.y, c = threadIdx.x;
  float o;
  if (c < 64) o = proj[(long long)j * CZ + c] * proj[(long long)i * CZ + 64 + c];
  else        o = proj[(long long)j * CZ + (c - 64)] - proj[(long long)i * CZ + 64 + (c - 64)];
  cat[((long long)i * L + j) * CZ + c] = o;
}
__global__ __launch_bounds__(128) void transpose_kernel(const float* __restrict__ x,
                                                        float* __restrict__ y) {
  const int i = blockIdx.x, j = blockIdx.y, c = threadIdx.x;
  y[((long long)j * L + i) * CZ + c] = x[((long long)i * L + j) * CZ + c];
}
__global__ __launch_bounds__(128) void transpose_add_kernel(float* __restrict__ y,
                                                            const float* __restrict__ x) {
  const int i = blockIdx.x, j = blockIdx.y, c = threadIdx.x;
  y[((long long)i * L + j) * CZ + c] += x[((long long)j * L + i) * CZ + c];
}

// ---------------------------------------------------------------------------
// Host-side launch helpers
// ---------------------------------------------------------------------------
static inline int log2i(int v) { int s = 0; while ((1 << s) < v) ++s; return s; }

static inline void launch_gemm(hipStream_t st, bool vec, const float* A, const float* B,
                               const float* bias, float* C, int M, int N, int K,
                               long long ldaM, long long ldaK,
                               long long ldbK, long long ldbN,
                               long long ldcM, long long ldcN,
                               int batchOuter, int batchInner,
                               long long sAo, long long sAi,
                               long long sBo, long long sBi,
                               long long sCo, long long sCi, int fuse) {
  GemmArgs g;
  g.A = A; g.B = B; g.bias = bias; g.C = C;
  g.M = M; g.N = N; g.K = K; g.batchShift = log2i(batchInner);
  g.ldaM = ldaM; g.ldaK = ldaK; g.ldbK = ldbK; g.ldbN = ldbN;
  g.ldcM = ldcM; g.ldcN = ldcN;
  g.sAo = sAo; g.sAi = sAi; g.sBo = sBo; g.sBi = sBi; g.sCo = sCo; g.sCi = sCi;
  g.fuse = fuse;
  dim3 grid((M + 63) / 64, (N + 63) / 64, batchOuter * batchInner);
  if (vec) wmma_gemm_kernel<1><<<grid, 128, 0, st>>>(g);
  else     wmma_gemm_kernel<0><<<grid, 128, 0, st>>>(g);
}
// plain row-major C[M,N] = A[M,K] @ W[K,N] (+bias, +act)   (fast VEC path)
static inline void gemm_rm(hipStream_t st, const float* A, const float* W,
                           const float* bias, float* C, int M, int N, int K,
                           int fuse) {
  launch_gemm(st, true, A, W, bias, C, M, N, K, K, 1, N, 1, N, 1,
              1, 1, 0, 0, 0, 0, 0, 0, fuse);
}
static inline void launch_ew(hipStream_t st, void (*k)(float*, const float*, long long),
                             float* y, const float* x, long long n) {
  int blocks = (int)((n + 255) / 256); if (blocks > 65535) blocks = 65535;
  k<<<blocks, 256, 0, st>>>(y, x, n);
}

// ---------------------------------------------------------------------------
// Triangle multiplication (outgoing/incoming)
// ---------------------------------------------------------------------------
static void run_trimul(hipStream_t st, float* z_io, bool incoming,
                       const float* ln_in_b, const float* ln_in_g,
                       const float* ag_b, const float* ag_w,
                       const float* ap_b, const float* ap_w,
                       const float* bg_b, const float* bg_w,
                       const float* bp_b, const float* bp_w,
                       const float* g_b,  const float* g_w,
                       const float* lno_b,const float* lno_g,
                       const float* o_b,  const float* o_w,
                       const float* pm, float* zn,
                       float* b1, float* b2, float* b3) {
  const long long LL = (long long)L * L;
  const long long NZ = LL * CZ;
  ln_kernel<<<(int)LL, 256, 0, st>>>(z_io, ln_in_g, ln_in_b, zn, CZ);
  // a = pm * sigmoid(zn@ag) * (zn@ap)
  gemm_rm(st, zn, ag_w, ag_b, b1, (int)LL, CZ, CZ, /*sigmoid*/2);
  gemm_rm(st, zn, ap_w, ap_b, b2, (int)LL, CZ, CZ, 0);
  { int blocks = (int)((NZ + 255) / 256); if (blocks > 65535) blocks = 65535;
    tri_gate_kernel<<<blocks, 256, 0, st>>>(b1, b1, b2, pm, NZ); }
  // b = pm * sigmoid(zn@bg) * (zn@bp)
  gemm_rm(st, zn, bg_w, bg_b, b2, (int)LL, CZ, CZ, 2);
  gemm_rm(st, zn, bp_w, bp_b, b3, (int)LL, CZ, CZ, 0);
  { int blocks = (int)((NZ + 255) / 256); if (blocks > 65535) blocks = 65535;
    tri_gate_kernel<<<blocks, 256, 0, st>>>(b2, b2, b3, pm, NZ); }
  // einsum per channel c: 128 batched [L x L x L] GEMMs (strided path)
  const long long LCZ = (long long)L * CZ;
  if (!incoming) {
    // x[i,j,c] = sum_k a[i,k,c] * b[j,k,c]
    launch_gemm(st, false, b1, b2, nullptr, b3, L, L, L,
                LCZ, CZ,            // A[i][k]
                CZ, LCZ,            // B[k][j] = b[j,k,c]
                LCZ, CZ,            // C[i][j]
                1, CZ, 0, 1, 0, 1, 0, 1, 0);
  } else {
    // x[i,j,c] = sum_k a[k,i,c] * b[k,j,c]
    launch_gemm(st, false, b1, b2, nullptr, b3, L, L, L,
                CZ, LCZ,            // A[i][k] = a[k,i,c]
                LCZ, CZ,            // B[k][j]
                LCZ, CZ,
                1, CZ, 0, 1, 0, 1, 0, 1, 0);
  }
  // x = ln_out(x);  z += (x@o + b) * sigmoid(zn@g + b)
  ln_kernel<<<(int)LL, 256, 0, st>>>(b3, lno_g, lno_b, b1, CZ);
  gemm_rm(st, zn, g_w, g_b, b2, (int)LL, CZ, CZ, 2);
  gemm_rm(st, b1, o_w, o_b, b3, (int)LL, CZ, CZ, 0);
  { int blocks = (int)((NZ + 255) / 256); if (blocks > 65535) blocks = 65535;
    addmul_kernel<<<blocks, 256, 0, st>>>(z_io, b3, b2, NZ); }
}

// ---------------------------------------------------------------------------
// Triangle attention (starting/ending). zsrc is z (or z^T for ending).
// ---------------------------------------------------------------------------
static void run_triattn(hipStream_t st, float* z_io, const float* zsrc, bool ending,
                        const float* ln_b, const float* ln_g,
                        const float* bias_w,
                        const float* q_w, const float* k_w, const float* v_w,
                        const float* g_b, const float* g_w,
                        const float* o_b, const float* o_w,
                        const float* pm, float* zn, float* tb,
                        float* b1, float* b2, float* b3, float* b4, float* chk) {
  const long long LL  = (long long)L * L;
  const long long NZ  = LL * CZ;
  const long long LCZ = (long long)L * CZ;
  const float scale = 0.17677669529663687f;   // 32^-0.5

  ln_kernel<<<(int)LL, 256, 0, st>>>(zsrc, ln_g, ln_b, zn, CZ);
  gemm_rm(st, zn, bias_w, nullptr, tb, (int)LL, HP, CZ, 0);   // [L,L,HP]
  gemm_rm(st, zn, q_w, nullptr, b1, (int)LL, CZ, CZ, 0);      // q
  gemm_rm(st, zn, k_w, nullptr, b2, (int)LL, CZ, CZ, 0);      // k
  gemm_rm(st, zn, v_w, nullptr, b3, (int)LL, CZ, CZ, 0);      // v
  gemm_rm(st, zn, g_w, g_b,     b4, (int)LL, CZ, CZ, 2);      // sigmoid gate

  for (int i0 = 0; i0 < L; i0 += CI) {
    // scores[(li,h)][q][k] = sum_d q[i,q,h,d] * k[i,k,h,d]   (B transposed)
    launch_gemm(st, false, b1 + (long long)i0 * LCZ, b2 + (long long)i0 * LCZ, nullptr, chk,
                L, L, HD,
                CZ, 1,              // A[q][d]
                1, CZ,              // B[d][k] = k[k][d]
                L, 1,
                CI, HP,
                LCZ, HD,            // A: outer i, inner h
                LCZ, HD,            // B
                (long long)HP * LL, LL, 0);
    tri_softmax_kernel<<<dim3(L, HP, CI), 256, 0, st>>>(chk, pm, tb, i0, scale);
    // o[i,q,h,d] = sum_k P[q][k] * v[i,k,h,d]   (overwrite q buffer with o)
    launch_gemm(st, true, chk, b3 + (long long)i0 * LCZ, nullptr, b1 + (long long)i0 * LCZ,
                L, HD, L,
                L, 1,               // A = P
                CZ, 1,              // B[k][d]
                CZ, 1,              // C[q][d]
                CI, HP,
                (long long)HP * LL, LL,
                LCZ, HD,
                LCZ, HD, 0);
  }
  // o *= gate ; out = o@Wo + bo ; z += out (transposed-add for "ending")
  launch_ew(st, mul_kernel, b1, b4, NZ);
  gemm_rm(st, b1, o_w, o_b, b3, (int)LL, CZ, CZ, 0);
  if (!ending) {
    launch_ew(st, add_inplace_kernel, z_io, b3, NZ);
  } else {
    transpose_add_kernel<<<dim3(L, L), 128, 0, st>>>(z_io, b3);
  }
}

// ---------------------------------------------------------------------------
// Entry point
// ---------------------------------------------------------------------------
extern "C" void kernel_launch(void* const* d_in, const int* in_sizes, int n_in,
                              void* d_out, int out_size, void* d_ws, size_t ws_size,
                              hipStream_t stream) {
  (void)in_sizes; (void)n_in; (void)out_size; (void)ws_size;
  const float* s_in = (const float*)d_in[0];
  const float* z_in = (const float*)d_in[1];
  const float* mask = (const float*)d_in[2];
  auto P = [&](int i) { return (const float*)d_in[i]; };

  // params flattened in JAX sorted-key pytree order, after (s, z, mask)
  const float *attn_g_b = P(3),  *attn_g_w = P(4);
  const float *attn_o_b = P(5),  *attn_o_w = P(6);
  const float *attn_qkv_w = P(7);
  const float *ln1_b = P(8),     *ln1_g = P(9);
  const float *mp_ln_b = P(10),  *mp_ln_g = P(11);
  const float *mp_w1_b = P(12),  *mp_w1_w = P(13);
  const float *mp_w2_b = P(14),  *mp_w2_w = P(15);
  const float *ms_ln_b = P(16),  *ms_ln_g = P(17);
  const float *ms_w1_b = P(18),  *ms_w1_w = P(19);
  const float *ms_w2_b = P(20),  *ms_w2_w = P(21);
  const float *p2s_w = P(22);
  const float *p2s_ln_b = P(23), *p2s_ln_g = P(24);
  const float *s2p_ln_b = P(25), *s2p_ln_g = P(26);
  const float *s2p_o_b = P(27),  *s2p_o_w = P(28);
  const float *s2p_pj_b = P(29), *s2p_pj_w = P(30);
  // tae: bias, g, k, ln, o, q, v
  const float *tae_bias_w = P(31);
  const float *tae_g_b = P(32), *tae_g_w = P(33);
  const float *tae_k_w = P(34);
  const float *tae_ln_b = P(35), *tae_ln_g = P(36);
  const float *tae_o_b = P(37),  *tae_o_w = P(38);
  const float *tae_q_w = P(39),  *tae_v_w = P(40);
  const float *tas_bias_w = P(41);
  const float *tas_g_b = P(42), *tas_g_w = P(43);
  const float *tas_k_w = P(44);
  const float *tas_ln_b = P(45), *tas_ln_g = P(46);
  const float *tas_o_b = P(47),  *tas_o_w = P(48);
  const float *tas_q_w = P(49),  *tas_v_w = P(50);
  // tmi/tmo: ag, ap, bg, bp, g, ln_in, ln_out, o
  const float *tmi_ag_b = P(51), *tmi_ag_w = P(52), *tmi_ap_b = P(53), *tmi_ap_w = P(54);
  const float *tmi_bg_b = P(55), *tmi_bg_w = P(56), *tmi_bp_b = P(57), *tmi_bp_w = P(58);
  const float *tmi_g_b  = P(59), *tmi_g_w  = P(60);
  const float *tmi_lni_b = P(61), *tmi_lni_g = P(62);
  const float *tmi_lno_b = P(63), *tmi_lno_g = P(64);
  const float *tmi_o_b = P(65),  *tmi_o_w = P(66);
  const float *tmo_ag_b = P(67), *tmo_ag_w = P(68), *tmo_ap_b = P(69), *tmo_ap_w = P(70);
  const float *tmo_bg_b = P(71), *tmo_bg_w = P(72), *tmo_bp_b = P(73), *tmo_bp_w = P(74);
  const float *tmo_g_b  = P(75), *tmo_g_w  = P(76);
  const float *tmo_lni_b = P(77), *tmo_lni_g = P(78);
  const float *tmo_lno_b = P(79), *tmo_lno_g = P(80);
  const float *tmo_o_b = P(81),  *tmo_o_w = P(82);

  float* s_out = (float*)d_out;                       // [L, CS]
  float* z_out = s_out + (long long)L * CS;           // [L, L, CZ]

  // workspace bump allocator
  const long long LL = (long long)L * L;
  const long long NZ = LL * CZ;
  float* w = (float*)d_ws;
  auto alloc = [&](long long n) { float* p = w; w += n; return p; };
  float* pm    = alloc(LL);
  float* sn    = alloc((long long)L * CS);
  float* tmp_s = alloc((long long)L * CS);
  float* gb_s  = alloc((long long)L * CS);
  float* ybuf  = alloc((long long)L * CS);
  float* qkv   = alloc((long long)L * 3 * CS);
  float* hseq  = alloc((long long)L * 4 * CS);
  float* biasS = alloc(LL * HS);
  float* scS   = alloc((long long)HS * LL);
  float* proj  = alloc((long long)L * CZ);
  float* tb    = alloc(LL * HP);
  float* zn    = alloc(NZ);
  float* b1    = alloc(NZ);
  float* b2    = alloc(NZ);
  float* b3    = alloc(NZ);
  float* b4    = alloc(NZ);
  float* zT    = alloc(NZ);
  float* chk   = alloc((long long)CI * HP * LL);  // >= RCHUNK*4*CZ too

  const float seq_scale = 0.17677669529663687f;  // (CS/HS)^-0.5

  // ---- pair mask ----
  { int blocks = (int)((LL + 255) / 256);
    pm_kernel<<<blocks, 256, 0, stream>>>(mask, pm); }

  // ---- PairToSequence bias = LN(z) @ p2s_w  -> [L,L,HS] ----
  ln_kernel<<<(int)LL, 256, 0, stream>>>(z_in, p2s_ln_g, p2s_ln_b, zn, CZ);
  gemm_rm(stream, zn, p2s_w, nullptr, biasS, (int)LL, HS, CZ, 0);

  // ---- sequence attention ----
  ln_kernel<<<L, 256, 0, stream>>>(s_in, ln1_g, ln1_b, sn, CS);
  gemm_rm(stream, sn, attn_qkv_w, nullptr, qkv, L, 3 * CS, CS, 0);  // [L,HS,96]
  // scores[h][q][k] = sum_d q[q,h,d]*k[k,h,d]    (B transposed -> strided path)
  launch_gemm(stream, false, qkv, qkv + HD, nullptr, scS, L, L, HD,
              3 * CS, 1,            // A[q][d]
              1, 3 * CS,            // B[d][k] = k[k][d]
              L, 1,
              1, HS, 0, 96, 0, 96, 0, LL, 0);
  seq_softmax_kernel<<<dim3(L, HS), 256, 0, stream>>>(scS, biasS, mask, seq_scale);
  // y[q,h,d] = sum_k P[h][q][k] * v[k,h,d]   (row-major -> VEC path)
  launch_gemm(stream, true, scS, qkv + 2 * HD, nullptr, ybuf, L, HD, L,
              L, 1,
              3 * CS, 1,
              CS, 1,
              1, HS, 0, LL, 0, 96, 0, HD, 0);
  gemm_rm(stream, sn, attn_g_w, attn_g_b, gb_s, L, CS, CS, /*sigmoid*/2);
  launch_ew(stream, mul_kernel, ybuf, gb_s, (long long)L * CS);
  gemm_rm(stream, ybuf, attn_o_w, attn_o_b, tmp_s, L, CS, CS, 0);
  { long long n = (long long)L * CS; int blocks = (int)((n + 255) / 256);
    add_kernel<<<blocks, 256, 0, stream>>>(s_out, s_in, tmp_s, n); }

  // ---- sequence transition ----
  ln_kernel<<<L, 256, 0, stream>>>(s_out, ms_ln_g, ms_ln_b, sn, CS);
  gemm_rm(stream, sn, ms_w1_w, ms_w1_b, hseq, L, 4 * CS, CS, /*relu*/1);
  gemm_rm(stream, hseq, ms_w2_w, ms_w2_b, tmp_s, L, CS, 4 * CS, 0);
  launch_ew(stream, add_inplace_kernel, s_out, tmp_s, (long long)L * CS);

  // ---- SequenceToPair ----
  ln_kernel<<<L, 256, 0, stream>>>(s_out, s2p_ln_g, s2p_ln_b, sn, CS);
  gemm_rm(stream, sn, s2p_pj_w, s2p_pj_b, proj, L, CZ, CS, 0);
  s2p_build_kernel<<<dim3(L, L), 128, 0, stream>>>(proj, b1);
  gemm_rm(stream, b1, s2p_o_w, s2p_o_b, b2, (int)LL, CZ, CZ, 0);
  { int blocks = (int)((NZ + 255) / 256); if (blocks > 65535) blocks = 65535;
    add_kernel<<<blocks, 256, 0, stream>>>(z_out, z_in, b2, NZ); }

  // ---- triangle multiplicative updates ----
  run_trimul(stream, z_out, /*incoming=*/false,
             tmo_lni_b, tmo_lni_g, tmo_ag_b, tmo_ag_w, tmo_ap_b, tmo_ap_w,
             tmo_bg_b, tmo_bg_w, tmo_bp_b, tmo_bp_w, tmo_g_b, tmo_g_w,
             tmo_lno_b, tmo_lno_g, tmo_o_b, tmo_o_w, pm, zn, b1, b2, b3);
  run_trimul(stream, z_out, /*incoming=*/true,
             tmi_lni_b, tmi_lni_g, tmi_ag_b, tmi_ag_w, tmi_ap_b, tmi_ap_w,
             tmi_bg_b, tmi_bg_w, tmi_bp_b, tmi_bp_w, tmi_g_b, tmi_g_w,
             tmi_lno_b, tmi_lno_g, tmi_o_b, tmi_o_w, pm, zn, b1, b2, b3);

  // ---- triangle attention (starting) ----
  run_triattn(stream, z_out, z_out, /*ending=*/false,
              tas_ln_b, tas_ln_g, tas_bias_w, tas_q_w, tas_k_w, tas_v_w,
              tas_g_b, tas_g_w, tas_o_b, tas_o_w, pm, zn, tb, b1, b2, b3, b4, chk);

  // ---- triangle attention (ending): operate on z^T, add back transposed ----
  transpose_kernel<<<dim3(L, L), 128, 0, stream>>>(z_out, zT);
  run_triattn(stream, z_out, zT, /*ending=*/true,
              tae_ln_b, tae_ln_g, tae_bias_w, tae_q_w, tae_k_w, tae_v_w,
              tae_g_b, tae_g_w, tae_o_b, tae_o_w, pm, zn, tb, b1, b2, b3, b4, chk);

  // ---- pair transition (chunked over rows: L^2 x CZ -> 4CZ -> CZ) ----
  ln_kernel<<<(int)LL, 256, 0, stream>>>(z_out, mp_ln_g, mp_ln_b, zn, CZ);
  for (long long r0 = 0; r0 < LL; r0 += RCHUNK) {
    gemm_rm(stream, zn + r0 * CZ, mp_w1_w, mp_w1_b, chk, RCHUNK, 4 * CZ, CZ, /*relu*/1);
    gemm_rm(stream, chk, mp_w2_w, mp_w2_b, b1, RCHUNK, CZ, 4 * CZ, 0);
    long long n = (long long)RCHUNK * CZ;
    int blocks = (int)((n + 255) / 256);
    add_inplace_kernel<<<blocks, 256, 0, stream>>>(z_out + r0 * CZ, b1, n);
  }
}
